// BatteryMoEFlattenIntraCycleMoELayer_51754355916892
// MI455X (gfx1250) — matrix operand order repaired and checked by reading, not compiled, not run on hardware
//
#include <hip/hip_runtime.h>
#include <stdint.h>

// Problem constants (from reference)
#define BB     128
#define LL     100
#define FF     900          // C*CURVE = 3*300
#define EE     8
#define DD     512
#define FP     928          // K padded to 29 * 32
#define MP     112          // M padded to 7 * 16
#define KTILES (FP / 32)    // 29
#define MTILES (MP / 16)    // 7

typedef __attribute__((ext_vector_type(16))) __bf16 v16bf;
typedef __attribute__((ext_vector_type(8)))  __bf16 v8bf;
typedef __attribute__((ext_vector_type(8)))  float  v8f;

__device__ __forceinline__ unsigned short f2bf(float f) {
    union { float f; unsigned int u; } c; c.f = f;
    unsigned int u = c.u;
    // round-to-nearest-even truncation to bf16
    unsigned int r = (u + 0x7FFFu + ((u >> 16) & 1u)) >> 16;
    return (unsigned short)r;
}

// ---------------- Kernel 1: gates (softmax -> mask -> top2 -> renorm) -------
__global__ void gates_kernel(const float* __restrict__ logits,
                             const int*   __restrict__ masks,
                             float* __restrict__ gsel,
                             int*   __restrict__ esel) {
    int b = blockIdx.x * blockDim.x + threadIdx.x;
    if (b >= BB) return;
    float l[EE];
    float mx = -3.4e38f;
    #pragma unroll
    for (int e = 0; e < EE; ++e) { l[e] = logits[b * EE + e]; mx = fmaxf(mx, l[e]); }
    float s = 0.f;
    float g[EE];
    #pragma unroll
    for (int e = 0; e < EE; ++e) { g[e] = __expf(l[e] - mx); s += g[e]; }
    #pragma unroll
    for (int e = 0; e < EE; ++e) { g[e] = (masks[b * EE + e] == 1) ? (g[e] / s) : 0.f; }
    // top-2, first-occurrence on ties (matches jax.lax.top_k)
    int i0 = 0; float v0 = g[0];
    #pragma unroll
    for (int e = 1; e < EE; ++e) { if (g[e] > v0) { v0 = g[e]; i0 = e; } }
    int i1 = -1; float v1 = -1.f;
    #pragma unroll
    for (int e = 0; e < EE; ++e) { if (e != i0 && g[e] > v1) { v1 = g[e]; i1 = e; } }
    float denom = v0 + v1 + 1e-9f;
    gsel[b * 2 + 0] = v0 / denom;
    gsel[b * 2 + 1] = v1 / denom;
    esel[b * 2 + 0] = i0;
    esel[b * 2 + 1] = i1;
}

// ---------------- Kernel 2: x fp32 -> bf16, padded [B][112][928] ------------
__global__ void convx_kernel(const float* __restrict__ x,
                             unsigned short* __restrict__ xb) {
    size_t idx = (size_t)blockIdx.x * blockDim.x + threadIdx.x;
    const size_t total = (size_t)BB * MP * FP;
    if (idx >= total) return;
    int    f = (int)(idx % FP);
    size_t t = idx / FP;
    int    m = (int)(t % MP);
    int    b = (int)(t / MP);
    float v = (m < LL && f < FF) ? x[((size_t)b * LL + m) * FF + f] : 0.f;
    xb[idx] = f2bf(v);
}

// ---------------- Kernel 3: W fp32 -> bf16, padded [E][512][928] ------------
__global__ void convw_kernel(const float* __restrict__ W,
                             unsigned short* __restrict__ wb) {
    size_t idx = (size_t)blockIdx.x * blockDim.x + threadIdx.x;
    const size_t total = (size_t)EE * DD * FP;
    if (idx >= total) return;
    int    f = (int)(idx % FP);
    size_t t = idx / FP;
    int    d = (int)(t % DD);
    int    e = (int)(t / DD);
    float v = (f < FF) ? W[((size_t)e * DD + d) * FF + f] : 0.f;
    wb[idx] = f2bf(v);
}

// ---------------- Kernel 4: top-2 fused WMMA GEMM ---------------------------
// grid = (BB*MTILES, 4), block = 256 (8 waves).
// Block (b, mt, ng): out rows [mt*16, mt*16+16), cols [ng*128, ng*128+128).
// Wave w owns N tile n0 = ng*128 + w*16; one shared A fragment feeds 2 WMMAs
// (one per selected expert) every K-step.
__global__ __launch_bounds__(256) void moe_wmma_kernel(
    const unsigned short* __restrict__ xb,   // bf16 bits [B][MP][FP]
    const unsigned short* __restrict__ wb,   // bf16 bits [E][DD][FP]
    const float* __restrict__ bias,          // fp32 [E][DD]
    const float* __restrict__ gsel,          // [B][2]
    const int*   __restrict__ esel,          // [B][2]
    unsigned short* __restrict__ out)        // bf16 bits [B][LL][DD]
{
    const int b    = blockIdx.x / MTILES;
    const int mt   = blockIdx.x % MTILES;
    const int wave = threadIdx.x >> 5;
    const int lane = threadIdx.x & 31;
    const int lrow = lane & 15;
    const int lhi  = lane >> 4;                 // 0: lanes 0-15, 1: lanes 16-31
    const int n0   = blockIdx.y * 128 + wave * 16;

    const float g0 = gsel[2 * b + 0];
    const float g1 = gsel[2 * b + 1];
    const int   e0 = esel[2 * b + 0];
    const int   e1 = esel[2 * b + 1];

    // A (16x32 bf16) lane layout: row M = lane%16; low lanes hold K 0..15,
    // high lanes K 16..31  -> two 16B chunks at k-offsets lhi*16 and lhi*16+8.
    const unsigned short* Arow = xb + ((size_t)(b * MP + mt * 16 + lrow)) * FP;
    const int aoff = lhi * 16;
    // B (32x16 bf16) lane layout: col N = lane%16; low lanes K 0..15, high lanes
    // K 16..31, contiguous -> one 32B v16bf load per lane from row-major W.
    const unsigned short* B0r = wb + ((size_t)(e0 * DD + n0 + lrow)) * FP + lhi * 16;
    const unsigned short* B1r = wb + ((size_t)(e1 * DD + n0 + lrow)) * FP + lhi * 16;

    v8f acc0 = {};
    v8f acc1 = {};
    union AU { v16bf v; v8bf h[2]; };

    for (int k = 0; k < KTILES; ++k) {
        const int kb = k * 32;
        AU a;
        a.h[0] = *(const v8bf*)(Arow + kb + aoff);
        a.h[1] = *(const v8bf*)(Arow + kb + aoff + 8);
        v16bf fb0 = *(const v16bf*)(B0r + kb);
        v16bf fb1 = *(const v16bf*)(B1r + kb);
        if (k + 4 < KTILES) {   // prefetch 4 K-tiles ahead (global_prefetch_b8)
            __builtin_prefetch((const void*)(Arow + kb + 128), 0, 1);
            __builtin_prefetch((const void*)(B0r  + kb + 128), 0, 1);
            __builtin_prefetch((const void*)(B1r  + kb + 128), 0, 1);
        }
        // 8 args: (neg_a, A, neg_b, B, c_mod, C, reuse_a, reuse_b)
        acc0 = __builtin_amdgcn_wmma_f32_16x16x32_bf16(
                   false, a.v, false, fb0, (short)0, acc0, false, false);
        acc1 = __builtin_amdgcn_wmma_f32_16x16x32_bf16(
                   false, a.v, false, fb1, (short)0, acc1, false, false);
    }

    // C/D layout: VGPR r -> row M = mt*16 + lhi*8 + r, col N = lane%16.
    const int d = n0 + lrow;
    const float bmix = g0 * bias[(size_t)e0 * DD + d] + g1 * bias[(size_t)e1 * DD + d];
    const int mbase = mt * 16 + lhi * 8;
    #pragma unroll
    for (int r = 0; r < 8; ++r) {
        int m = mbase + r;
        if (m < LL) {
            float v = g0 * acc0[r] + g1 * acc1[r] + bmix;
            out[((size_t)b * LL + m) * DD + d] = f2bf(v);
        }
    }
}

// ---------------------------------------------------------------------------
extern "C" void kernel_launch(void* const* d_in, const int* in_sizes, int n_in,
                              void* d_out, int out_size, void* d_ws, size_t ws_size,
                              hipStream_t stream) {
    const float* x      = (const float*)d_in[0];   // [B,L,C,CURVE] fp32
    const float* logits = (const float*)d_in[1];   // [B,E] fp32
    const int*   masks  = (const int*)  d_in[2];   // [B,E] int32
    const float* W      = (const float*)d_in[3];   // [E,D,F] fp32
    const float* bias   = (const float*)d_in[4];   // [E,D] fp32
    unsigned short* out = (unsigned short*)d_out;  // [B,L,D] bf16 bits

    char* ws = (char*)d_ws;
    const size_t xb_elems = (size_t)BB * MP * FP;  // 13,303,808
    const size_t wb_elems = (size_t)EE * DD * FP;  //  3,801,088
    unsigned short* xb = (unsigned short*)ws;
    unsigned short* wb = (unsigned short*)(ws + xb_elems * 2);
    float* gsel = (float*)(ws + xb_elems * 2 + wb_elems * 2);
    int*   esel = (int*)((char*)gsel + (size_t)BB * 2 * sizeof(float));

    gates_kernel<<<1, 128, 0, stream>>>(logits, masks, gsel, esel);

    {
        size_t total = xb_elems;
        convx_kernel<<<(unsigned)((total + 255) / 256), 256, 0, stream>>>(x, xb);
    }
    {
        size_t total = wb_elems;
        convw_kernel<<<(unsigned)((total + 255) / 256), 256, 0, stream>>>(W, wb);
    }

    dim3 grid(BB * MTILES, DD / 128);   // (896, 4)
    moe_wmma_kernel<<<grid, 256, 0, stream>>>(xb, wb, bias, gsel, esel, out);
}
